// PCTVQVAE_57097295233301
// MI455X (gfx1250) — matrix operand order repaired and checked by reading, not compiled, not run on hardware
//
#include <hip/hip_runtime.h>
#include <hip/hip_bf16.h>
#include <math.h>

typedef __attribute__((ext_vector_type(16))) __bf16 bf16x16;
typedef __attribute__((ext_vector_type(8)))  __bf16 bf16x8;
typedef __attribute__((ext_vector_type(8)))  float  f32x8;

#define BATCH 4096
#define JT    24
#define IND   9
#define HD    512
#define TI    64
#define NBLK  4
#define MTOK  32
#define KC    2048
#define DD    512
#define NTOK  (BATCH*MTOK)   /* 131072 */
#define NROW  (BATCH*JT)     /* 98304  */

__device__ __forceinline__ __bf16 f2bf(float f){
  unsigned u = __builtin_bit_cast(unsigned, f);
  unsigned r = u + 0x7FFFu + ((u >> 16) & 1u);           // RNE truncate
  unsigned short h = (unsigned short)(r >> 16);
  return __builtin_bit_cast(__bf16, h);
}

__device__ __forceinline__ float bf2f(__bf16 b){
  unsigned short s = __builtin_bit_cast(unsigned short, b);
  unsigned u = (unsigned)s << 16;
  return __builtin_bit_cast(float, u);
}

__device__ __forceinline__ float gelu_exact(float x){
  return 0.5f * x * (1.0f + erff(x * 0.70710678118654752440f));
}

__device__ __forceinline__ f32x8 wmma_bf16(bf16x16 a, bf16x16 b, f32x8 c){
  return __builtin_amdgcn_wmma_f32_16x16x32_bf16(false, a, false, b, (short)0, c, false, false);
}

// ---- WMMA fragment loaders (contiguous k = 16*(lane/16) + s, wave32) -------------------

// A: 16x32 bf16 tile from LDS (row-major, leading dim lda). Two b128 loads.
__device__ __forceinline__ bf16x16 frag_a_lds(const __bf16* A, int lda, int row0, int k0, int lane){
  const __bf16* p = A + (row0 + (lane & 15)) * lda + k0 + ((lane >> 4) << 4);
  bf16x8 lo = *(const bf16x8*)p;
  bf16x8 hi = *(const bf16x8*)(p + 8);
  return __builtin_shufflevector(lo, hi, 0,1,2,3,4,5,6,7,8,9,10,11,12,13,14,15);
}

// B: 32x16 tile from a PRE-PACKED bf16 matrix stored [N][K] (ldk = DD).
// lane gives N; its 16 K-values are contiguous => one aligned 32B load.
__device__ __forceinline__ bf16x16 frag_b_pk(const __bf16* Wt, int k0, int n0, int lane){
  const __bf16* p = Wt + (size_t)(n0 + (lane & 15)) * DD + k0 + ((lane >> 4) << 4);
  return *(const bf16x16*)p;
}

// ---- weight packing: f32 [K][N=512] -> bf16 [N][K=512] --------------------

__global__ __launch_bounds__(256) void k_packT(const float* __restrict__ W, __bf16* __restrict__ Wt){
  int i = blockIdx.x * 256 + threadIdx.x;     // over 512*512
  int k = i & (HD - 1);
  int n = i >> 9;
  Wt[(size_t)n * HD + k] = f2bf(W[(size_t)k * HD + n]);
}

__global__ __launch_bounds__(256) void k_packcb(const float* __restrict__ CB, __bf16* __restrict__ out){
  size_t i = (size_t)blockIdx.x * 256 + threadIdx.x;
  if (i < (size_t)KC * DD) out[i] = f2bf(CB[i]);
}

// ---- small elementwise / small-K kernels ----------------------------------

__global__ __launch_bounds__(256) void k_start(const float* __restrict__ pose,
    const float* __restrict__ w, const float* __restrict__ b, float* __restrict__ X){
  size_t i = (size_t)blockIdx.x * 256 + threadIdx.x;
  if (i >= (size_t)NROW * HD) return;
  int h = (int)(i % HD);
  size_t row = i / HD;
  const float* pr = pose + row * IND;
  float acc = b[h];
#pragma unroll
  for (int t = 0; t < IND; ++t) acc += pr[t] * w[t * HD + h];
  X[i] = acc;
}

__global__ __launch_bounds__(256) void k_tokproj(const float* __restrict__ X,
    const float* __restrict__ w, const float* __restrict__ b, float* __restrict__ Y){
  size_t i = (size_t)blockIdx.x * 256 + threadIdx.x;
  if (i >= (size_t)NTOK * HD) return;
  int h = (int)(i % HD);
  int m = (int)((i / HD) % MTOK);
  size_t bb = i / ((size_t)HD * MTOK);
  const float* xp = X + bb * JT * HD + h;
  float acc = b[m];
#pragma unroll
  for (int j = 0; j < JT; ++j) acc += xp[(size_t)j * HD] * w[j * MTOK + m];
  Y[i] = acc;
}

__global__ __launch_bounds__(256) void k_dectok(const float* __restrict__ CB,
    const int* __restrict__ idx, const float* __restrict__ w, const float* __restrict__ b,
    float* __restrict__ Y){
  size_t i = (size_t)blockIdx.x * 256 + threadIdx.x;
  if (i >= (size_t)NROW * DD) return;
  int d = (int)(i % DD);
  int j = (int)((i / DD) % JT);
  size_t bb = i / ((size_t)DD * JT);
  const int* ip = idx + bb * MTOK;
  float acc = b[j];
  for (int m = 0; m < MTOK; ++m) acc += CB[(size_t)ip[m] * DD + d] * w[m * JT + j];
  Y[i] = acc;
}

__global__ __launch_bounds__(256) void k_rec(const float* __restrict__ Y,
    const float* __restrict__ w, const float* __restrict__ b, float* __restrict__ out){
  size_t i = (size_t)blockIdx.x * 256 + threadIdx.x;
  if (i >= (size_t)NROW * IND) return;
  int o = (int)(i % IND);
  size_t row = i / IND;
  const float* yp = Y + row * HD;
  float acc = b[o];
  for (int h = 0; h < HD; ++h) acc += yp[h] * w[h * IND + o];
  out[i] = acc;
}

__global__ __launch_bounds__(256) void k_ln_rows(float* __restrict__ X,
    const float* __restrict__ sc, const float* __restrict__ bi, int rows){
  int row = blockIdx.x * 8 + (threadIdx.x >> 5);
  if (row >= rows) return;
  int lane = threadIdx.x & 31;
  float* p = X + (size_t)row * HD;
  float s = 0.f, s2 = 0.f;
  for (int c = lane; c < HD; c += 32){ float v = p[c]; s += v; s2 += v * v; }
#pragma unroll
  for (int o = 16; o > 0; o >>= 1){ s += __shfl_xor(s, o, 32); s2 += __shfl_xor(s2, o, 32); }
  float mu = s * (1.f / HD), var = s2 * (1.f / HD) - mu * mu, rs = rsqrtf(var + 1e-5f);
  for (int c = lane; c < HD; c += 32) p[c] = (p[c] - mu) * rs * sc[c] + bi[c];
}

// ---- fused mixer block (one workgroup per batch element, in-place) --------
// LDS: xf(48K) + yf(48K) + 64K bf16 arena shared between token-mix hidden
// ([64][512]) and the WMMA A/H panels (2 x [32][512]).  Total 160 KB.

__global__ __launch_bounds__(256) void k_mixer(float* __restrict__ X,
    const float* __restrict__ ln1_s, const float* __restrict__ ln1_b,
    const float* __restrict__ tw1,  const float* __restrict__ tb1,
    const float* __restrict__ tw2,  const float* __restrict__ tb2,
    const float* __restrict__ ln2_s, const float* __restrict__ ln2_b,
    const __bf16* __restrict__ cw1t, const float* __restrict__ cb1,
    const __bf16* __restrict__ cw2t, const float* __restrict__ cb2){
  __shared__ float  xf[JT][HD];          // input (residual)
  __shared__ float  yf[JT][HD];          // LN1 out, then token-mix out
  __shared__ __attribute__((aligned(32))) __bf16 arena[TI * HD];   // 64 KB aliased

  __bf16* hid = arena;                   // [TI][HD]  token-mix hidden
  __bf16* abf = arena;                   // [32][HD]  WMMA A panel
  __bf16* hbf = arena + 32 * HD;         // [32][HD]  gelu hidden panel

  const int tid = threadIdx.x, lane = tid & 31, wv = tid >> 5;
  float* gx = X + (size_t)blockIdx.x * JT * HD;

  for (int i = tid; i < JT * HD; i += 256) (&xf[0][0])[i] = gx[i];
  __syncthreads();

  // LN1 -> yf
  for (int r = wv; r < JT; r += 8){
    float s = 0.f, s2 = 0.f;
    for (int c = lane; c < HD; c += 32){ float v = xf[r][c]; s += v; s2 += v * v; }
#pragma unroll
    for (int o = 16; o > 0; o >>= 1){ s += __shfl_xor(s, o, 32); s2 += __shfl_xor(s2, o, 32); }
    float mu = s * (1.f / HD), var = s2 * (1.f / HD) - mu * mu, rs = rsqrtf(var + 1e-5f);
    for (int c = lane; c < HD; c += 32) yf[r][c] = (xf[r][c] - mu) * rs * ln1_s[c] + ln1_b[c];
  }
  __syncthreads();

  // token-mix pass A: hid[i][h] = gelu( sum_j yf[j][h]*tw1[j,i] + tb1[i] )
  for (int e = tid; e < TI * HD; e += 256){
    int h = e & (HD - 1);
    int i = e >> 9;
    float acc = tb1[i];
#pragma unroll 8
    for (int j = 0; j < JT; ++j) acc += yf[j][h] * tw1[j * TI + i];
    hid[i * HD + h] = f2bf(gelu_exact(acc));
  }
  __syncthreads();

  // token-mix pass B: yf[j][h] = sum_i hid[i][h]*tw2[i,j] + tb2[j]
  for (int e = tid; e < JT * HD; e += 256){
    int h = e & (HD - 1);
    int j = e >> 9;
    float acc = tb2[j];
#pragma unroll 8
    for (int i = 0; i < TI; ++i) acc += bf2f(hid[i * HD + h]) * tw2[i * JT + j];
    yf[j][h] = acc;
  }
  __syncthreads();

  // LN2(x + y) -> abf rows 0..23 ; rows 24..31 zeroed (arena reuse)
  for (int i = tid; i < 8 * HD; i += 256) abf[24 * HD + i] = f2bf(0.f);
  for (int r = wv; r < JT; r += 8){
    float s = 0.f, s2 = 0.f;
    for (int c = lane; c < HD; c += 32){ float v = xf[r][c] + yf[r][c]; s += v; s2 += v * v; }
#pragma unroll
    for (int o = 16; o > 0; o >>= 1){ s += __shfl_xor(s, o, 32); s2 += __shfl_xor(s2, o, 32); }
    float mu = s * (1.f / HD), var = s2 * (1.f / HD) - mu * mu, rs = rsqrtf(var + 1e-5f);
    for (int c = lane; c < HD; c += 32){
      float v = xf[r][c] + yf[r][c];
      abf[r * HD + c] = f2bf((v - mu) * rs * ln2_s[c] + ln2_b[c]);
    }
  }
  __syncthreads();

  // channel GEMM1: hbf = gelu(abf @ cw1 + cb1), [32x512] @ [512x512]
  for (int t8 = wv; t8 < 64; t8 += 8){
    int mt = t8 >> 5, nt = t8 & 31;
    f32x8 acc = {};
#pragma unroll 4
    for (int kk = 0; kk < 16; ++kk){
      bf16x16 a = frag_a_lds(abf, HD, mt * 16, kk * 32, lane);
      bf16x16 b = frag_b_pk(cw1t, kk * 32, nt * 16, lane);
      acc = wmma_bf16(a, b, acc);
    }
    int col = nt * 16 + (lane & 15);
    int rb  = mt * 16 + ((lane >> 4) << 3);
    float bias = cb1[col];
#pragma unroll
    for (int r = 0; r < 8; ++r) hbf[(rb + r) * HD + col] = f2bf(gelu_exact(acc[r] + bias));
  }
  __syncthreads();

  // channel GEMM2 + residual: X = x + y + (hbf @ cw2 + cb2)
  for (int t8 = wv; t8 < 64; t8 += 8){
    int mt = t8 >> 5, nt = t8 & 31;
    f32x8 acc = {};
#pragma unroll 4
    for (int kk = 0; kk < 16; ++kk){
      bf16x16 a = frag_a_lds(hbf, HD, mt * 16, kk * 32, lane);
      bf16x16 b = frag_b_pk(cw2t, kk * 32, nt * 16, lane);
      acc = wmma_bf16(a, b, acc);
    }
    int col = nt * 16 + (lane & 15);
    int rb  = mt * 16 + ((lane >> 4) << 3);
    float bias = cb2[col];
#pragma unroll
    for (int r = 0; r < 8; ++r){
      int row = rb + r;
      if (row < JT) gx[row * HD + col] = acc[r] + bias + xf[row][col] + yf[row][col];
    }
  }
}

// ---- generic M x 512 x 512 GEMM + bias (WMMA, packed bf16 weights) --------

__global__ __launch_bounds__(256) void k_gemm512(const float* __restrict__ A,
    const __bf16* __restrict__ Wt, const float* __restrict__ bias, float* __restrict__ C){
  __shared__ __attribute__((aligned(32))) __bf16 As[64][HD];   // 64 KB
  const int tid = threadIdx.x, lane = tid & 31, wv = tid >> 5;
  size_t row0 = (size_t)blockIdx.x * 64;
  int col0 = blockIdx.y * 64;
  for (int i = tid; i < 64 * HD; i += 256) (&As[0][0])[i] = f2bf(A[row0 * HD + i]);
  __syncthreads();
  for (int t = wv; t < 16; t += 8){
    int mt = t >> 2, nt = t & 3;
    f32x8 acc = {};
#pragma unroll 4
    for (int kk = 0; kk < 16; ++kk){
      bf16x16 a = frag_a_lds(&As[0][0], HD, mt * 16, kk * 32, lane);
      bf16x16 b = frag_b_pk(Wt, kk * 32, col0 + nt * 16, lane);
      acc = wmma_bf16(a, b, acc);
    }
    int col = col0 + nt * 16 + (lane & 15);
    size_t rowb = row0 + mt * 16 + ((lane >> 4) << 3);
    float bv = bias[col];
#pragma unroll
    for (int r = 0; r < 8; ++r) C[(rowb + r) * HD + col] = acc[r] + bv;
  }
}

// ---- VQ: codebook norms, argmin via WMMA, EMA scatter, finalize -----------

__global__ __launch_bounds__(256) void k_cnorm(const float* __restrict__ CB, float* __restrict__ cn){
  int k = blockIdx.x;
  const float* p = CB + (size_t)k * DD;
  float s = 0.f;
  for (int d = threadIdx.x; d < DD; d += 256){ float v = p[d]; s += v * v; }
  int lane = threadIdx.x & 31, wv = threadIdx.x >> 5;
#pragma unroll
  for (int o = 16; o > 0; o >>= 1) s += __shfl_xor(s, o, 32);
  __shared__ float red[8];
  if (lane == 0) red[wv] = s;
  __syncthreads();
  if (threadIdx.x == 0){ float t = 0.f; for (int i = 0; i < 8; ++i) t += red[i]; cn[k] = t; }
}

__global__ __launch_bounds__(256) void k_vq(const float* __restrict__ E,
    const __bf16* __restrict__ CBt, const float* __restrict__ cn, int* __restrict__ idx_out){
  __shared__ __attribute__((aligned(32))) __bf16 Es[64][DD];   // 64 KB
  __shared__ float sc[64][32];                                 // 8 KB score tile
  const int tid = threadIdx.x, lane = tid & 31, wv = tid >> 5;
  size_t row0 = (size_t)blockIdx.x * 64;
  for (int i = tid; i < 64 * DD; i += 256) (&Es[0][0])[i] = f2bf(E[row0 * DD + i]);
  __syncthreads();
  float bestv = 3.4e38f; int besti = 0;
  int mt = wv & 3, nt = wv >> 2;                  // 8 waves -> 4x2 tile grid (64x32)
  for (int c0 = 0; c0 < KC; c0 += 32){
    f32x8 acc = {};
#pragma unroll 4
    for (int kk = 0; kk < 16; ++kk){
      bf16x16 a = frag_a_lds(&Es[0][0], DD, mt * 16, kk * 32, lane);
      bf16x16 b = frag_b_pk(CBt, kk * 32, c0 + nt * 16, lane);   // codebook is [code][d] == [N][K]
      acc = wmma_bf16(a, b, acc);
    }
    int cl = nt * 16 + (lane & 15);
    int rb = mt * 16 + ((lane >> 4) << 3);
    float cnv = cn[c0 + cl];
    __syncthreads();                              // previous tile fully consumed
#pragma unroll
    for (int r = 0; r < 8; ++r) sc[rb + r][cl] = cnv - 2.f * acc[r];
    __syncthreads();
    if (tid < 64){
      for (int c = 0; c < 32; ++c){ float v = sc[tid][c]; if (v < bestv){ bestv = v; besti = c0 + c; } }
    }
  }
  if (tid < 64) idx_out[row0 + tid] = besti;
}

__global__ __launch_bounds__(256) void k_ema_loss(const float* __restrict__ E,
    const float* __restrict__ CB, const int* __restrict__ idx,
    float* __restrict__ counts, float* __restrict__ dw, float* __restrict__ scal){
  int n = blockIdx.x;
  int k = idx[n];
  if (threadIdx.x == 0) atomicAdd(&counts[k], 1.f);
  float s = 0.f;
  for (int d = threadIdx.x; d < DD; d += 256){
    float e = E[(size_t)n * DD + d];
    float q = CB[(size_t)k * DD + d];
    float df = q - e; s += df * df;
    atomicAdd(&dw[(size_t)k * DD + d], e);
  }
  int lane = threadIdx.x & 31, wv = threadIdx.x >> 5;
#pragma unroll
  for (int o = 16; o > 0; o >>= 1) s += __shfl_xor(s, o, 32);
  __shared__ float red[8];
  if (lane == 0) red[wv] = s;
  __syncthreads();
  if (threadIdx.x == 0){ float t = 0.f; for (int i = 0; i < 8; ++i) t += red[i]; atomicAdd(&scal[1], t); }
}

__global__ __launch_bounds__(256) void k_ema_cs(const float* __restrict__ oldcs,
    const float* __restrict__ counts, float* __restrict__ csnew, float* __restrict__ scal){
  float s = 0.f;
  for (int k = threadIdx.x; k < KC; k += 256){
    float v = oldcs[k] * 0.9f + 0.1f * counts[k];
    csnew[k] = v; s += v;
  }
  int lane = threadIdx.x & 31, wv = threadIdx.x >> 5;
#pragma unroll
  for (int o = 16; o > 0; o >>= 1) s += __shfl_xor(s, o, 32);
  __shared__ float red[8];
  if (lane == 0) red[wv] = s;
  __syncthreads();
  if (threadIdx.x == 0){ float t = 0.f; for (int i = 0; i < 8; ++i) t += red[i]; scal[0] = t; }
}

__global__ __launch_bounds__(256) void k_newcb(const float* __restrict__ emw,
    const float* __restrict__ dw, const float* __restrict__ csnew,
    const float* __restrict__ scal, float* __restrict__ out){
  size_t i = (size_t)blockIdx.x * 256 + threadIdx.x;
  if (i >= (size_t)KC * DD) return;
  int k = (int)(i / DD);
  float n = scal[0];
  float cs = (csnew[k] + 1e-5f) / (n + KC * 1e-5f) * n;
  out[i] = (0.9f * emw[i] + 0.1f * dw[i]) / cs;
}

__global__ void k_loss_fin(const float* __restrict__ scal, float* __restrict__ out){
  out[0] = scal[1] * (1.f / ((float)NTOK * (float)DD));
}

// ---- driver ---------------------------------------------------------------

extern "C" void kernel_launch(void* const* d_in, const int* in_sizes, int n_in,
                              void* d_out, int out_size, void* d_ws, size_t ws_size,
                              hipStream_t stream){
  (void)in_sizes; (void)n_in; (void)out_size; (void)ws_size;
  const float* pose     = (const float*)d_in[0];
  const float* start_w  = (const float*)d_in[1];
  const float* start_b  = (const float*)d_in[2];
  const float* enc_ln_s = (const float*)d_in[51];
  const float* enc_ln_b = (const float*)d_in[52];
  const float* tok_w    = (const float*)d_in[53];
  const float* tok_b    = (const float*)d_in[54];
  const float* feat_w   = (const float*)d_in[55];
  const float* feat_b   = (const float*)d_in[56];
  const float* codebook = (const float*)d_in[57];
  const float* ema_cs   = (const float*)d_in[58];
  const float* ema_w    = (const float*)d_in[59];
  const float* dtw      = (const float*)d_in[60];
  const float* dtb      = (const float*)d_in[61];
  const float* dsw      = (const float*)d_in[62];
  const float* dsb      = (const float*)d_in[63];
  const float* dec_ln_s = (const float*)d_in[112];
  const float* dec_ln_b = (const float*)d_in[113];
  const float* rec_w    = (const float*)d_in[114];
  const float* rec_b    = (const float*)d_in[115];

  float* ws = (float*)d_ws;
  size_t o = 0;
  float* xbuf   = ws + o; o += (size_t)NROW * HD;   // encoder, then decoder-token buf
  float* tokbuf = ws + o; o += (size_t)NTOK * HD;   // token-proj / decoder activations
  float* ebuf   = ws + o; o += (size_t)NTOK * DD;   // e (kept for EMA + loss)
  float* cnorm  = ws + o; o += KC;
  float* counts = ws + o; o += KC;
  float* dwb    = ws + o; o += (size_t)KC * DD;
  float* csnew  = ws + o; o += KC;
  float* scal   = ws + o; o += 8;                   // [0]=n, [1]=loss accum
  __bf16* packw = (__bf16*)(ws + o); o += (size_t)18 * HD * HD / 2;  // 18 packed 512x512 bf16
  __bf16* cbbf  = (__bf16*)(ws + o); o += (size_t)KC * DD / 2;       // packed codebook bf16

  float* outf     = (float*)d_out;
  float* rec_out  = outf;                                    // B*J*9
  int*   idx_out  = (int*)(outf + (size_t)NROW * IND);       // NTOK ints
  float* loss_out = outf + (size_t)NROW * IND + NTOK;        // 1
  float* cb_out   = loss_out + 1;                            // K*D

  (void)hipMemsetAsync(counts, 0, KC * sizeof(float), stream);
  (void)hipMemsetAsync(dwb, 0, (size_t)KC * DD * sizeof(float), stream);
  (void)hipMemsetAsync(scal, 0, 8 * sizeof(float), stream);

  dim3 blk(256);
  const unsigned gpack = (HD * HD) / 256;   // 1024

  // ---- pack weights once: bf16, [N][K] ----
  // slots: 2i / 2i+1 = enc[i] cw1/cw2 ; 8+2i / 9+2i = dec[i] cw1/cw2 ; 16 = feat ; 17 = dec_start
  for (int i = 0; i < NBLK; ++i){
    void* const* pe = d_in + 3 + 12 * i;
    void* const* pd = d_in + 64 + 12 * i;
    k_packT<<<gpack, blk, 0, stream>>>((const float*)pe[8],  packw + (size_t)(2*i)   * HD * HD);
    k_packT<<<gpack, blk, 0, stream>>>((const float*)pe[10], packw + (size_t)(2*i+1) * HD * HD);
    k_packT<<<gpack, blk, 0, stream>>>((const float*)pd[8],  packw + (size_t)(8+2*i) * HD * HD);
    k_packT<<<gpack, blk, 0, stream>>>((const float*)pd[10], packw + (size_t)(9+2*i) * HD * HD);
  }
  k_packT<<<gpack, blk, 0, stream>>>(feat_w, packw + (size_t)16 * HD * HD);
  k_packT<<<gpack, blk, 0, stream>>>(dsw,    packw + (size_t)17 * HD * HD);
  k_packcb<<<(unsigned)(((size_t)KC * DD + 255) / 256), blk, 0, stream>>>(codebook, cbbf);

  // ---- encoder ----
  k_start<<<(unsigned)(((size_t)NROW * HD + 255) / 256), blk, 0, stream>>>(pose, start_w, start_b, xbuf);
  for (int i = 0; i < NBLK; ++i){
    void* const* p = d_in + 3 + 12 * i;
    k_mixer<<<BATCH, blk, 0, stream>>>(xbuf,
        (const float*)p[0], (const float*)p[1], (const float*)p[2], (const float*)p[3],
        (const float*)p[4], (const float*)p[5], (const float*)p[6], (const float*)p[7],
        packw + (size_t)(2*i) * HD * HD, (const float*)p[9],
        packw + (size_t)(2*i+1) * HD * HD, (const float*)p[11]);
  }
  k_ln_rows<<<NROW / 8, blk, 0, stream>>>(xbuf, enc_ln_s, enc_ln_b, NROW);
  k_tokproj<<<(unsigned)(((size_t)NTOK * HD + 255) / 256), blk, 0, stream>>>(xbuf, tok_w, tok_b, tokbuf);
  { dim3 g(NTOK / 64, HD / 64); k_gemm512<<<g, blk, 0, stream>>>(tokbuf, packw + (size_t)16 * HD * HD, feat_b, ebuf); }

  // ---- VQ ----
  k_cnorm<<<KC, blk, 0, stream>>>(codebook, cnorm);
  k_vq<<<NTOK / 64, blk, 0, stream>>>(ebuf, cbbf, cnorm, idx_out);
  k_ema_loss<<<NTOK, blk, 0, stream>>>(ebuf, codebook, idx_out, counts, dwb, scal);
  k_ema_cs<<<1, blk, 0, stream>>>(ema_cs, counts, csnew, scal);
  k_newcb<<<(unsigned)(((size_t)KC * DD + 255) / 256), blk, 0, stream>>>(ema_w, dwb, csnew, scal, cb_out);
  k_loss_fin<<<1, 1, 0, stream>>>(scal, loss_out);

  // ---- decoder (straight-through forward value == quantized codes) ----
  k_dectok<<<(unsigned)(((size_t)NROW * DD + 255) / 256), blk, 0, stream>>>(codebook, idx_out, dtw, dtb, xbuf);
  { dim3 g(NROW / 64, HD / 64); k_gemm512<<<g, blk, 0, stream>>>(xbuf, packw + (size_t)17 * HD * HD, dsb, tokbuf); }
  for (int i = 0; i < NBLK; ++i){
    void* const* p = d_in + 64 + 12 * i;
    k_mixer<<<BATCH, blk, 0, stream>>>(tokbuf,
        (const float*)p[0], (const float*)p[1], (const float*)p[2], (const float*)p[3],
        (const float*)p[4], (const float*)p[5], (const float*)p[6], (const float*)p[7],
        packw + (size_t)(8+2*i) * HD * HD, (const float*)p[9],
        packw + (size_t)(9+2*i) * HD * HD, (const float*)p[11]);
  }
  k_ln_rows<<<NROW / 8, blk, 0, stream>>>(tokbuf, dec_ln_s, dec_ln_b, NROW);
  k_rec<<<(unsigned)(((size_t)NROW * IND + 255) / 256), blk, 0, stream>>>(tokbuf, rec_w, rec_b, rec_out);
}
